// BasicBlock_704374637356
// MI455X (gfx1250) — compile-verified, matrix operand order checked by reading, not run on hardware
//
#include <hip/hip_runtime.h>
#include <math.h>

// ---------------------------------------------------------------------------
// CDNA5 (gfx1250) wave32 WMMA helpers. Fixed geometry: 256x256 images, B=8.
// ---------------------------------------------------------------------------
typedef __attribute__((ext_vector_type(16))) _Float16 v16h;
typedef __attribute__((ext_vector_type(8)))  _Float16 v8h;
typedef __attribute__((ext_vector_type(8)))  float    v8f;

#define WMMA_F32_F16(a, b, c) \
    __builtin_amdgcn_wmma_f32_16x16x32_f16(false, (a), false, (b), (short)0, (c), false, false)

#define NPX 65536       // pixels per image (256*256)
#define NBA 8           // batch

// ISA 7.12.2: 16-bit A (16x32): lanes0-15 hold K 0-7 & 16-23 of row M=lane;
// lanes16-31 hold K 8-15 & 24-31 of row M=lane-16. Same k-pattern for B, col=lane&15.
__device__ __forceinline__ int kmap(int lane, int j) {
    return ((j < 8) ? j : j + 8) + ((lane & 16) ? 8 : 0);
}
__device__ __forceinline__ v16h pack16(v8h lo, v8h hi) {
    v16h r;
#pragma unroll
    for (int j = 0; j < 8; ++j) { r[j] = lo[j]; r[j + 8] = hi[j]; }
    return r;
}
// Build a fragment from 16 k-contiguous f32 (row-resident in memory): 4x b128 loads.
__device__ __forceinline__ v16h frag_row_f32(const float* row, int half) {
    const float4* r0 = (const float4*)(row + half * 8);
    const float4* r1 = (const float4*)(row + 16 + half * 8);
    float4 a0 = r0[0], a1 = r0[1], b0 = r1[0], b1 = r1[1];
    v16h f;
    f[0]  = (_Float16)a0.x; f[1]  = (_Float16)a0.y; f[2]  = (_Float16)a0.z; f[3]  = (_Float16)a0.w;
    f[4]  = (_Float16)a1.x; f[5]  = (_Float16)a1.y; f[6]  = (_Float16)a1.z; f[7]  = (_Float16)a1.w;
    f[8]  = (_Float16)b0.x; f[9]  = (_Float16)b0.y; f[10] = (_Float16)b0.z; f[11] = (_Float16)b0.w;
    f[12] = (_Float16)b1.x; f[13] = (_Float16)b1.y; f[14] = (_Float16)b1.z; f[15] = (_Float16)b1.w;
    return f;
}
__device__ __forceinline__ float gelu_f(float x) {
    return 0.5f * x * (1.0f + erff(x * 0.7071067811865475f));
}

// ---------------------------------------------------------------------------
// Per-pixel LayerNorm over C channels. Exact grid: B*NPX threads.
// ---------------------------------------------------------------------------
template <int C>
__global__ void ln_t(const float* __restrict__ in, float* __restrict__ out,
                     const float* __restrict__ w, const float* __restrict__ b)
{
    unsigned idx = blockIdx.x * blockDim.x + threadIdx.x;   // < B*NPX
    unsigned bb = idx >> 16;
    unsigned p  = idx & 65535u;
    const float* ip = in  + (size_t)bb * C * NPX + p;
    float*       op = out + (size_t)bb * C * NPX + p;
    float vals[C];
    float mu = 0.f;
#pragma unroll
    for (int c = 0; c < C; ++c) { vals[c] = ip[(size_t)c * NPX]; mu += vals[c]; }
    mu *= (1.0f / C);
    float var = 0.f;
#pragma unroll
    for (int c = 0; c < C; ++c) { float d = vals[c] - mu; var += d * d; }
    var *= (1.0f / C);
    float rs = rsqrtf(var + 1e-5f);
#pragma unroll
    for (int c = 0; c < C; ++c)
        op[(size_t)c * NPX] = (vals[c] - mu) * rs * w[c] + b[c];
}

// ---------------------------------------------------------------------------
// 1x1 conv (channel GEMM) via WMMA. Branchless fragments; exact grid:
// blocks*8 waves == B*4096 tiles.
// ---------------------------------------------------------------------------
template <int CIN, int COUT>
__global__ void conv1x1_t(const float* __restrict__ in,
                          float* __restrict__ out,
                          const float* __restrict__ W, const float* __restrict__ bias)
{
    int lane = threadIdx.x & 31, wave = threadIdx.x >> 5;
    unsigned tile = blockIdx.x * 8u + wave;                 // < B*4096
    unsigned b       = tile >> 12;
    unsigned pixBase = (tile & 4095u) << 4;
    const float* inp  = in  + (size_t)b * CIN * NPX;
    float*       outp = out + (size_t)b * COUT * NPX;
    int m = lane & 15;
    v16h a;
#pragma unroll
    for (int j = 0; j < 16; ++j) {
        int k  = kmap(lane, j);
        int kc = (k < CIN) ? k : (CIN - 1);   // clamped address; B holds the zero
        a[j] = (_Float16)inp[(size_t)kc * NPX + pixBase + m];
    }
    const int NT = (COUT + 15) >> 4;
    int mo = (lane & 16) ? 8 : 0;
#pragma unroll
    for (int t = 0; t < NT; ++t) {
        int co  = (t << 4) + (lane & 15);
        int coc = (co < COUT) ? co : (COUT - 1);
        v16h bf;
#pragma unroll
        for (int j = 0; j < 16; ++j) {
            int k  = kmap(lane, j);
            int kc = (k < CIN) ? k : (CIN - 1);
            bf[j] = (_Float16)W[coc * CIN + kc];
        }
        if (CIN < 32) {                        // zero padded-k element (one cndmask)
            bf[15] = (lane & 16) ? (_Float16)0.f : bf[15];
        }
        v8f c = {};
        c = WMMA_F32_F16(a, bf, c);
        if (co < COUT) {
            float bv = bias ? bias[co] : 0.f;
#pragma unroll
            for (int r = 0; r < 8; ++r)
                outp[(size_t)co * NPX + pixBase + r + mo] = c[r] + bv;
        }
    }
}

// concat([x(1ch), z(31ch)]) -> 1x1 conv 32->32 via WMMA (all dims exact, branch-free)
__global__ void concatconv_wmma_kernel(const float* __restrict__ xc, const float* __restrict__ z31,
                                       const float* __restrict__ W, const float* __restrict__ bias,
                                       float* __restrict__ out)
{
    int lane = threadIdx.x & 31, wave = threadIdx.x >> 5;
    unsigned tile = blockIdx.x * 8u + wave;
    unsigned b       = tile >> 12;
    unsigned pixBase = (tile & 4095u) << 4;
    int m = lane & 15;
    v16h a;
#pragma unroll
    for (int j = 0; j < 16; ++j) {
        int k = kmap(lane, j);
        const float* src = (k == 0) ? (xc + (size_t)b * NPX)
                                    : (z31 + ((size_t)b * 31 + (k - 1)) * NPX);
        a[j] = (_Float16)src[pixBase + m];
    }
    int mo = (lane & 16) ? 8 : 0;
#pragma unroll
    for (int t = 0; t < 2; ++t) {
        int co = (t << 4) + (lane & 15);
        v16h bf;
#pragma unroll
        for (int j = 0; j < 16; ++j) bf[j] = (_Float16)W[co * 32 + kmap(lane, j)];
        v8f c = {};
        c = WMMA_F32_F16(a, bf, c);
        float bv = bias[co];
#pragma unroll
        for (int r = 0; r < 8; ++r)
            out[((size_t)b * 32 + co) * NPX + pixBase + r + mo] = c[r] + bv;
    }
}

// ---------------------------------------------------------------------------
// Depthwise 3x3, zero padding, optional exact-gelu. Exact grid: B*C*NPX threads.
// ---------------------------------------------------------------------------
template <int C, int GELU>
__global__ void dw3x3_t(const float* __restrict__ in, float* __restrict__ out,
                        const float* __restrict__ W9, const float* __restrict__ bias)
{
    unsigned idx = blockIdx.x * blockDim.x + threadIdx.x;   // < B*C*NPX
    unsigned p  = idx & 65535u;
    unsigned bc = idx >> 16;
    unsigned c  = bc % (unsigned)C;                          // constant mod -> mulhi
    int y = p >> 8, x = p & 255;
    const float* ip = in + ((size_t)bc << 16);
    const float* wr = W9 + c * 9;
    float acc = bias ? bias[c] : 0.f;
#pragma unroll
    for (int dy = -1; dy <= 1; ++dy)
#pragma unroll
        for (int dx = -1; dx <= 1; ++dx) {
            int yy = y + dy, xx = x + dx;
            if (yy >= 0 && yy < 256 && xx >= 0 && xx < 256)
                acc += wr[(dy + 1) * 3 + (dx + 1)] * ip[(yy << 8) + xx];
        }
    out[idx] = GELU ? gelu_f(acc) : acc;
}

// x1 branch: LN over 1 channel is analytically = nl_lnx_b, so x1 is closed-form.
// Exact grid: 31*NPX threads.
__global__ void x1_kernel(float* __restrict__ x1, const float* __restrict__ lnxb,
                          const float* __restrict__ p1w, const float* __restrict__ p1b,
                          const float* __restrict__ p2w, const float* __restrict__ p2b)
{
    unsigned idx = blockIdx.x * blockDim.x + threadIdx.x;   // < 31*NPX
    unsigned p = idx & 65535u;
    unsigned c = idx >> 16;
    int y = p >> 8, x = p & 255;
    float cc = p1w[c] * lnxb[0] + p1b[c];
    float s = 0.f;
#pragma unroll
    for (int dy = -1; dy <= 1; ++dy)
#pragma unroll
        for (int dx = -1; dx <= 1; ++dx) {
            int yy = y + dy, xx = x + dx;
            if (yy >= 0 && yy < 256 && xx >= 0 && xx < 256)
                s += p2w[c * 9 + (dy + 1) * 3 + (dx + 1)];
        }
    x1[idx] = cc * s + p2b[c];
}

// per-(b,c) L2 norm over NPX pixels; one workgroup per row; C=31 compile-time
__global__ void rownorm_kernel(const float* __restrict__ in, long batchStride,
                               float* __restrict__ norms)
{
    unsigned row = blockIdx.x;
    unsigned b = row / 31u, c = row % 31u;                   // constant div
    const float* p = in + (size_t)b * batchStride + (size_t)c * NPX;
    float s = 0.f;
    for (int i = threadIdx.x; i < NPX; i += 256) { float v = p[i]; s += v * v; }
    __shared__ float red[256];
    red[threadIdx.x] = s;
    __syncthreads();
    for (int st = 128; st > 0; st >>= 1) {
        if ((int)threadIdx.x < st) red[threadIdx.x] += red[threadIdx.x + st];
        __syncthreads();
    }
    if (threadIdx.x == 0) norms[row] = sqrtf(red[0]);
}

__global__ void zero_kernel(float* __restrict__ p)
{
    p[blockIdx.x * blockDim.x + threadIdx.x] = 0.f;          // exact grid
}

// ---------------------------------------------------------------------------
// Gram matrix G[b,c,d] = sum_n q[b,c,n]*k[b,d,n] via WMMA over n-chunks of 32.
// Branchless: rows clamped (junk rows/cols land in the 32x32 pad), per-lane
// k-contiguous data as 4x b128. grid = B * 4 * 16 slices, 8 waves each.
// ---------------------------------------------------------------------------
__global__ void gram_wmma_kernel(const float* __restrict__ q, long qBS,
                                 const float* __restrict__ kk, long kBS,
                                 float* __restrict__ G, int C)
{
    int lane = threadIdx.x & 31, wave = threadIdx.x >> 5;
    unsigned id = blockIdx.x;
    unsigned slice = id & 15u; id >>= 4;
    unsigned t = id & 3u; unsigned b = id >> 2;
    int mt = t >> 1, ntl = t & 1;
    int half = (lane & 16) ? 1 : 0;
    int cRow = (mt  << 4) + (lane & 15);
    int dCol = (ntl << 4) + (lane & 15);
    int cR = (cRow < C) ? cRow : (C - 1);
    int dC = (dCol < C) ? dCol : (C - 1);
    const float* qrow = q  + (size_t)b * qBS + (size_t)cR * NPX;
    const float* krow = kk + (size_t)b * kBS + (size_t)dC * NPX;
    v8f acc = {};
    // 2048 chunks of 32 pixels; 16 slices * 8 waves -> 16 chunks per wave
#pragma unroll 4
    for (int ci = wave; ci < 128; ci += 8) {
        int n0 = ((int)slice * 128 + ci) << 5;
        v16h a  = frag_row_f32(qrow + n0, half);
        v16h bf = frag_row_f32(krow + n0, half);
        acc = WMMA_F32_F16(a, bf, acc);
    }
    int mo = (lane & 16) ? 8 : 0;
#pragma unroll
    for (int r = 0; r < 8; ++r) {
        int c = (mt << 4) + r + mo;   // pad rows/cols (>=C) are in-bounds and unused
        atomicAdd(&G[(size_t)b * 1024 + c * 32 + dCol], acc[r]);
    }
}

// softmax over d of G/(|q_c||k_d|) then fuse M = Wout @ att; writes zero-padded 32x32
__global__ void softmax_fuse_kernel(const float* __restrict__ G,
                                    const float* __restrict__ qn, const float* __restrict__ kn,
                                    int knBatched, const float* __restrict__ Wout,
                                    float* __restrict__ M)
{
    const int C = 31;
    int b = blockIdx.x;
    __shared__ float att[32 * 32];
    if ((int)threadIdx.x < C) {
        int c = threadIdx.x;
        float qv = fmaxf(qn[b * C + c], 1e-12f);
        float row[32];
        float mx = -1e30f;
        for (int d = 0; d < C; ++d) {
            float kv = fmaxf(kn[(knBatched ? b * C : 0) + d], 1e-12f);
            float s = G[(size_t)b * 1024 + c * 32 + d] / (qv * kv);
            row[d] = s; mx = fmaxf(mx, s);
        }
        float sum = 0.f;
        for (int d = 0; d < C; ++d) { row[d] = expf(row[d] - mx); sum += row[d]; }
        float inv = 1.f / sum;
        for (int d = 0; d < C; ++d) att[c * 32 + d] = row[d] * inv;
    }
    __syncthreads();
    for (int i = threadIdx.x; i < 1024; i += blockDim.x) {
        int c = i >> 5, d = i & 31;
        float s = 0.f;
        if (c < C && d < C)
            for (int e = 0; e < C; ++e) s += Wout[c * C + e] * att[e * 32 + d];
        M[(size_t)b * 1024 + i] = s;   // zero padding in rows/cols >= C
    }
}

// out[b,c,p] = sum_d Mat[b,c,d]*V[b,d,p] + bias[c] + res + (extra?)  via WMMA.
// Mat is zero-padded 32x32 -> unconditional 4x b128 fragment loads; V row clamped.
__global__ void applymat_wmma_kernel(const float* __restrict__ Mat,
                                     const float* __restrict__ V, long vBS,
                                     const float* __restrict__ bias,
                                     const float* __restrict__ res,
                                     const float* __restrict__ extra,
                                     float* __restrict__ out)
{
    const int C = 31;
    int lane = threadIdx.x & 31, wave = threadIdx.x >> 5;
    unsigned tile = blockIdx.x * 8u + wave;
    unsigned b       = tile >> 12;
    unsigned pixBase = (tile & 4095u) << 4;
    unsigned pix = pixBase + (lane & 15);
    v16h bf;
#pragma unroll
    for (int j = 0; j < 16; ++j) {
        int d  = kmap(lane, j);
        int dc = (d < C) ? d : (C - 1);   // Mat pad col is zero
        bf[j] = (_Float16)V[(size_t)b * vBS + (size_t)dc * NPX + pix];
    }
    int mo = (lane & 16) ? 8 : 0;
    int half = (lane & 16) ? 1 : 0;
#pragma unroll
    for (int mt = 0; mt < 2; ++mt) {
        int c0 = (mt << 4) + (lane & 15);
        v16h a = frag_row_f32(Mat + (size_t)b * 1024 + c0 * 32, half);
        v8f c = {};
        c = WMMA_F32_F16(a, bf, c);
#pragma unroll
        for (int r = 0; r < 8; ++r) {
            int cc = (mt << 4) + r + mo;
            if (cc < C) {
                size_t off = (size_t)b * 31 * NPX + (size_t)cc * NPX + pix;
                float v = c[r] + bias[cc] + res[off];
                if (extra) v += extra[off];
                out[off] = v;
            }
        }
    }
}

// ---------------------------------------------------------------------------
// Fully fused FFN block:
//   out = W3 @ gelu(dw3x3(gelu(W1 @ LN(xin)))) + xin + (res2?)
// 16x16 pixel tile + 1 halo; hidden (128ch, f16) lives entirely in LDS.
// LDS: lnT[324][32]f16 | wlds 4096 f16 (W1 then W3) | h1[324][128]f16 | h2[256][128]f16
// ---------------------------------------------------------------------------
#define LNT_OFF  0
#define WL_OFF   20736
#define H1_OFF   (20736 + 8192)
#define H2_OFF   (20736 + 8192 + 82944)
#define FFN_SMEM (20736 + 8192 + 82944 + 65536)   // 177408 B < 320 KB WGP LDS

__launch_bounds__(256)
__global__ void ffn_wmma_kernel(const float* __restrict__ xin, const float* __restrict__ res2,
                                float* __restrict__ out,
                                const float* __restrict__ lnw, const float* __restrict__ lnb,
                                const float* __restrict__ W1, const float* __restrict__ W2,
                                const float* __restrict__ W3)
{
    extern __shared__ char smem[];
    _Float16* lnT  = (_Float16*)(smem + LNT_OFF);
    _Float16* wlds = (_Float16*)(smem + WL_OFF);
    _Float16* h1   = (_Float16*)(smem + H1_OFF);
    _Float16* h2   = (_Float16*)(smem + H2_OFF);
    int b = blockIdx.x >> 8, tileId = blockIdx.x & 255;
    int y0 = (tileId >> 4) << 4, x0 = (tileId & 15) << 4;
    const float* xb = xin + (size_t)b * 32 * NPX;
    int tid = threadIdx.x, lane = tid & 31, wave = tid >> 5;
    int half = (lane & 16) ? 1 : 0;
    int mo = half ? 8 : 0;

    // W1 (f32 [128][32]) -> f16 LDS, same layout
    for (int i = tid; i < 4096; i += 256) wlds[i] = (_Float16)W1[i];

    // LayerNorm halo tile (zero outside image; W1 has no bias so h1(pad)=gelu(0)=0)
    for (int p = tid; p < 324; p += 256) {
        int gy = y0 + p / 18 - 1, gx = x0 + p % 18 - 1;
        if (gy >= 0 && gy < 256 && gx >= 0 && gx < 256) {
            size_t base = (size_t)gy * 256 + gx;
            float vals[32], mu = 0.f;
#pragma unroll
            for (int c = 0; c < 32; ++c) { vals[c] = xb[(size_t)c * NPX + base]; mu += vals[c]; }
            mu *= 0.03125f;
            float var = 0.f;
#pragma unroll
            for (int c = 0; c < 32; ++c) { float d = vals[c] - mu; var += d * d; }
            var *= 0.03125f;
            float rs = rsqrtf(var + 1e-5f);
#pragma unroll
            for (int c = 0; c < 32; ++c)
                lnT[p * 32 + c] = (_Float16)((vals[c] - mu) * rs * lnw[c] + lnb[c]);
        } else {
#pragma unroll
            for (int c = 0; c < 32; ++c) lnT[p * 32 + c] = (_Float16)0.f;
        }
    }
    __syncthreads();

    // h1 = gelu(W1 @ ln): 21 M-tiles x 8 N-tiles, K=32 in a single WMMA each
    for (int job = wave; job < 21 * 8; job += 8) {
        int mt = job >> 3, nt = job & 7;
        int p  = (mt << 4) + (lane & 15);
        v16h a = {};
        if (p < 324) {
            const v8h* rp = (const v8h*)(lnT + p * 32);
            a = pack16(rp[half], rp[2 + half]);
        }
        int co = (nt << 4) + (lane & 15);
        const v8h* wp = (const v8h*)(wlds + co * 32);
        v16h bf = pack16(wp[half], wp[2 + half]);
        v8f c = {};
        c = WMMA_F32_F16(a, bf, c);
#pragma unroll
        for (int r = 0; r < 8; ++r) {
            int pp = (mt << 4) + r + mo;
            if (pp < 324) h1[pp * 128 + co] = (_Float16)gelu_f(c[r]);
        }
    }
    __syncthreads();

    // W3 (f32 [32][128]) -> f16 LDS (reuse wlds region)
    for (int i = tid; i < 4096; i += 256) wlds[i] = (_Float16)W3[i];

    // h2 = gelu(dw3x3(h1)) on the 256 interior pixels
    for (int i = tid; i < 256 * 128; i += 256) {
        int ch = i & 127, p = i >> 7;
        int py = p >> 4, px = p & 15;
        float acc = 0.f;
#pragma unroll
        for (int dy = 0; dy < 3; ++dy)
#pragma unroll
            for (int dx = 0; dx < 3; ++dx)
                acc += W2[ch * 9 + dy * 3 + dx] *
                       (float)h1[((py + dy) * 18 + (px + dx)) * 128 + ch];
        h2[p * 128 + ch] = (_Float16)gelu_f(acc);
    }
    __syncthreads();

    // out = W3 @ h2 + xin + res2 : 16 M-tiles x 2 N-tiles, K=128 (4 chained WMMAs)
    for (int job = wave; job < 16 * 2; job += 8) {
        int mt = job >> 1, nt = job & 1;
        int p  = (mt << 4) + (lane & 15);
        int co = (nt << 4) + (lane & 15);
        v8f c = {};
#pragma unroll
        for (int kc = 0; kc < 4; ++kc) {
            const v8h* rp = (const v8h*)(h2 + p * 128 + kc * 32);
            v16h a = pack16(rp[half], rp[2 + half]);
            const v8h* wp = (const v8h*)(wlds + co * 128 + kc * 32);
            v16h bf = pack16(wp[half], wp[2 + half]);
            c = WMMA_F32_F16(a, bf, c);
        }
#pragma unroll
        for (int r = 0; r < 8; ++r) {
            int pp = (mt << 4) + r + mo;
            size_t gi = (size_t)(y0 + (pp >> 4)) * 256 + (x0 + (pp & 15));
            size_t oidx = (size_t)b * 32 * NPX + (size_t)co * NPX + gi;
            float v = c[r] + xb[(size_t)co * NPX + gi];
            if (res2) v += res2[oidx];
            out[oidx] = v;
        }
    }
}

// ---------------------------------------------------------------------------
// Host orchestration
// ---------------------------------------------------------------------------
extern "C" void kernel_launch(void* const* d_in, const int* in_sizes, int n_in,
                              void* d_out, int out_size, void* d_ws, size_t ws_size,
                              hipStream_t stream)
{
    (void)in_sizes; (void)n_in; (void)out_size; (void)ws_size;
    const int  N = NPX, B = NBA;
    const long S31 = 31L * N, S62 = 62L * N;
#define PRM(i) ((const float*)d_in[i])
    const float* x     = PRM(0);
    const float* z_pre = PRM(1);
    const float* z_cur = PRM(2);

    float* ws = (float*)d_ws;
    const size_t P31 = (size_t)B * 31 * N;
    const size_t P62 = (size_t)B * 62 * N;
    float* p1  = ws;                 // preln -> z (stage A result)
    float* p2  = p1 + P31;           // curln -> q -> z0 -> z2
    float* p3  = p2 + P31;           // qtmp/t/g/petmp -> out_x
    float* p4  = p3 + P31;           // kvtmp (62ch) ; later z1 (p4a) / x_forward
    float* p4b = p4 + P31;           // pe
    float* p5  = p4 + P62;           // kv (62ch); later x_input (32ch)
    float* x1b = p5 + P62;           // x1 plane [31][N] (batch-invariant)
    float* qn  = x1b + (size_t)31 * N;
    float* kn  = qn + 256;
    float* z1n = kn + 256;
    float* xn  = z1n + 256;
    float* G   = xn + 64;            // 8*32*32
    float* Mf  = G + 8192;
    float* G2  = Mf + 8192;
    float* M2  = G2 + 8192;

    const int blk = 256;
    const int gridLN   = (B * N) / 256;                      // exact
    const int gridGEMM = (B * (N >> 4)) / 8;                 // 4096 (8 waves/wg)
    const int gridDW31 = (int)((size_t)B * 31 * N / 256);    // exact
    const int gridDW62 = (int)((size_t)B * 62 * N / 256);    // exact
    const int gridX1   = 31 * N / 256;                       // exact
    const int gridGRAM = B * 4 * 16;

    // ---- Stage A: channel attention between z_pre and z_cur ----
    ln_t<31><<<gridLN, blk, 0, stream>>>(z_pre, p1, PRM(3), PRM(4));
    ln_t<31><<<gridLN, blk, 0, stream>>>(z_cur, p2, PRM(5), PRM(6));
    conv1x1_t<31, 31><<<gridGEMM, blk, 0, stream>>>(p2, p3, PRM(7), PRM(8));
    dw3x3_t<31, 0><<<gridDW31, blk, 0, stream>>>(p3, p2, PRM(9), PRM(10));     // q -> p2
    conv1x1_t<31, 62><<<gridGEMM, blk, 0, stream>>>(p1, p4, PRM(11), PRM(12));
    dw3x3_t<62, 0><<<gridDW62, blk, 0, stream>>>(p4, p5, PRM(13), PRM(14));    // kv -> p5
    rownorm_kernel<<<B * 31, blk, 0, stream>>>(p2, S31, qn);                   // |q|
    rownorm_kernel<<<B * 31, blk, 0, stream>>>(p5, S62, kn);                   // |k|
    zero_kernel<<<32, blk, 0, stream>>>(G);
    gram_wmma_kernel<<<gridGRAM, blk, 0, stream>>>(p2, S31, p5, S62, G, 31);
    softmax_fuse_kernel<<<B, blk, 0, stream>>>(G, qn, kn, 1, PRM(15), Mf);     // W_out@att
    applymat_wmma_kernel<<<gridGEMM, blk, 0, stream>>>(Mf, p5 + S31 /*v*/, S62, PRM(16),
                                                       z_cur, nullptr, p1);    // z -> p1

    // ---- Stage B: blockNL cross-attention ----
    ln_t<31><<<gridLN, blk, 0, stream>>>(p1, p2, PRM(19), PRM(20));            // z0 -> p2
    conv1x1_t<31, 31><<<gridGEMM, blk, 0, stream>>>(p2, p3, PRM(21), PRM(22));
    dw3x3_t<31, 0><<<gridDW31, blk, 0, stream>>>(p3, p4, PRM(23), PRM(24));    // z1 -> p4
    x1_kernel<<<gridX1, blk, 0, stream>>>(x1b, PRM(18), PRM(25), PRM(26), PRM(27), PRM(28));
    rownorm_kernel<<<B * 31, blk, 0, stream>>>(p4, S31, z1n);
    rownorm_kernel<<<31, blk, 0, stream>>>(x1b, 0, xn);
    zero_kernel<<<32, blk, 0, stream>>>(G2);
    gram_wmma_kernel<<<gridGRAM, blk, 0, stream>>>(p4, S31, x1b, 0, G2, 31);
    softmax_fuse_kernel<<<B, blk, 0, stream>>>(G2, z1n, xn, 0, PRM(33), M2);   // nl_w@att2
    conv1x1_t<31, 31><<<gridGEMM, blk, 0, stream>>>(p2, p3, PRM(29), PRM(30));
    dw3x3_t<31, 0><<<gridDW31, blk, 0, stream>>>(p3, p2, PRM(31), PRM(32));    // z2 -> p2
    dw3x3_t<31, 1><<<gridDW31, blk, 0, stream>>>(p2, p3, PRM(37), nullptr);    // gelu(dw)
    dw3x3_t<31, 0><<<gridDW31, blk, 0, stream>>>(p3, p4b, PRM(38), nullptr);   // pe -> p4b
    applymat_wmma_kernel<<<gridGEMM, blk, 0, stream>>>(M2, p2, S31, PRM(34),
                                                       p1, p4b, p3);           // out_x -> p3
    concatconv_wmma_kernel<<<gridGEMM, blk, 0, stream>>>(x, p3, PRM(35), PRM(36), p5); // x_input -> p5

    // ---- Stage C: two fused FFN blocks (hidden 128ch stays in LDS) ----
    (void)hipFuncSetAttribute((const void*)ffn_wmma_kernel,
                              hipFuncAttributeMaxDynamicSharedMemorySize, FFN_SMEM);
    ffn_wmma_kernel<<<B * 256, blk, FFN_SMEM, stream>>>(p5, nullptr, p4,
                                                        PRM(39), PRM(40), PRM(43), PRM(44), PRM(45));
    // final = x_input + x_backward = body(LN(x_forward)) + x_forward + x_input
    ffn_wmma_kernel<<<B * 256, blk, FFN_SMEM, stream>>>(p4, p5, (float*)d_out,
                                                        PRM(41), PRM(42), PRM(46), PRM(47), PRM(48));
}
